// ThalamicGating_41626823033590
// MI455X (gfx1250) — compile-verified, matrix operand order checked by reading, not run on hardware
//
#include <hip/hip_runtime.h>
#include <hip/hip_bf16.h>

// ---------------------------------------------------------------------------
// ThalamicGating fused kernel for gfx1250 (MI455X)
//   GEMM1(1024->512) + ReLU + GEMM2(512->1024) + sigmoid + gate + top-256
// Memory-bound (~0.65 GB HBM vs 0.275 TFLOP): single fused pass over the
// features (f32 parked in LDS, bf16 copy for WMMA), bf16 v_wmma_f32_16x16x32.
// kt-outer / nt-inner loop order keeps the register working set at
// ~64 VGPRs of accumulators + 24 VGPRs of fragments -> no scratch spills.
// ---------------------------------------------------------------------------

typedef __attribute__((ext_vector_type(16))) __bf16 v16bf;
typedef __attribute__((ext_vector_type(8)))  float  v8f;

union Frag {
    uint4  u[2];   // two 16-byte chunks (ds_load_b128 / global_load_b128)
    v16bf  v;      // 16 bf16 elements per lane
};

#define DIN  1024
#define DHID 512
#define KSEL 256
#define ROWS 32          // rows per workgroup (2 M-tiles)
#define KT1  32          // DIN/32  k-steps in GEMM1
#define KT2  16          // DHID/32 k-steps in GEMM2

static __device__ __forceinline__ unsigned int f2bf(float f) {
    unsigned int u = __float_as_uint(f);
    unsigned int r = u + 0x7FFFu + ((u >> 16) & 1u);   // round-to-nearest-even
    return r >> 16;
}

// ---------------------------------------------------------------------------
// Kernel 1: convert W1 [1024x512] and W2 [512x1024] (f32, row-major) to bf16,
// pre-swizzled into per-(n_tile,k_tile) WMMA B-fragment lane order:
//   frag[(nt*KT + kt)*512 + lane*16 + e], lane half selects K subblock.
// ---------------------------------------------------------------------------
__global__ void swizzle_weights_kernel(const float* __restrict__ W1,
                                       const float* __restrict__ W2,
                                       unsigned short* __restrict__ W1s,
                                       unsigned short* __restrict__ W2s) {
    const int b   = blockIdx.x;     // 0..1023 -> W1 frags, 1024..2047 -> W2 frags
    const int tid = threadIdx.x;

    for (int i = tid; i < 512; i += 256) {
        const int lane = i >> 4, e = i & 15;
        const int half = lane >> 4, idx = lane & 15;
        const int kk = (e < 8) ? (half * 8 + e) : (16 + half * 8 + (e - 8));
        if (b < 1024) {                       // W1 fragment: nt in [0,32), kt in [0,32)
            const int nt = b >> 5, kt = b & 31;
            const int n = nt * 16 + idx;
            const int k = kt * 32 + kk;
            W1s[(size_t)b * 512 + i] = (unsigned short)f2bf(W1[(size_t)k * DHID + n]);
        } else {                              // W2 fragment: nt in [0,64), kt in [0,16)
            const int b2 = b - 1024;
            const int nt = b2 >> 4, kt = b2 & 15;
            const int n = nt * 16 + idx;
            const int k = kt * 32 + kk;
            W2s[(size_t)b2 * 512 + i] = (unsigned short)f2bf(W2[(size_t)k * DIN + n]);
        }
    }
}

// ---------------------------------------------------------------------------
// Kernel 2: fused gate MLP + top-K per 32-row tile. 256 threads = 8 waves.
// LDS: 64 KB bf16 feats + 32 KB bf16 h + 128 KB f32 gated = 224 KB / WGP.
// ---------------------------------------------------------------------------
__global__ __launch_bounds__(256) void
thalamic_gate_topk_kernel(const float* __restrict__ feats,
                          const float* __restrict__ b1,
                          const float* __restrict__ b2,
                          const unsigned short* __restrict__ W1s,
                          const unsigned short* __restrict__ W2s,
                          float* __restrict__ out) {
    __shared__ alignas(16) unsigned short sFeatB[ROWS * DIN];   //  64 KB bf16 feats
    __shared__ alignas(16) unsigned short sH[ROWS * DHID];      //  32 KB bf16 hidden
    __shared__ alignas(16) float          sGated[ROWS * DIN];   // 128 KB f32 feats->gated

    const int tid  = threadIdx.x;
    const int lane = tid & 31;
    const int wave = tid >> 5;
    const int half = lane >> 4;
    const int idx  = lane & 15;
    const int rowBase = blockIdx.x * ROWS;

    // ---- Stage 0: stream features ONCE; park f32 in sGated, bf16 in sFeatB --
    for (int i = 0; i < 32; ++i) {
        const int v4 = i * 256 + tid;         // 0..8191 float4 slots (128 rows x 64)
        const int r  = v4 >> 6;
        const int c4 = v4 & 63;
        const float4 f4 =
            reinterpret_cast<const float4*>(feats + (size_t)(rowBase + r) * DIN)[c4];
        reinterpret_cast<float4*>(&sGated[r * DIN])[c4] = f4;
        const unsigned int lo = f2bf(f4.x) | (f2bf(f4.y) << 16);
        const unsigned int hi = f2bf(f4.z) | (f2bf(f4.w) << 16);
        *reinterpret_cast<uint2*>(&sFeatB[r * DIN + c4 * 4]) = make_uint2(lo, hi);
    }
    __syncthreads();

    // ---- Stage 1: h = relu(X @ W1 + b1) --------------------------------
    // kt outer / nt inner: acc[4 nt][2 M] = 64 VGPRs; A frags loaded once
    // per kt from LDS and reused by 4 B fragments (8 WMMAs per kt).
    {
        v8f acc[8] = {};                                 // [nti*2 + mtile]
        for (int kt = 0; kt < KT1; ++kt) {
            Frag a0, a1;
            const uint4* pa0 =
                reinterpret_cast<const uint4*>(&sFeatB[idx * DIN + kt * 32 + half * 8]);
            a0.u[0] = pa0[0];        // K = kt*32 + half*8 + [0..7]
            a0.u[1] = pa0[2];        // K = kt*32 + 16 + half*8 + [0..7]
            const uint4* pa1 = reinterpret_cast<const uint4*>(
                &sFeatB[(16 + idx) * DIN + kt * 32 + half * 8]);
            a1.u[0] = pa1[0];
            a1.u[1] = pa1[2];
            #pragma unroll
            for (int nti = 0; nti < 4; ++nti) {
                const int nt = wave * 4 + nti;           // 0..31
                Frag bfr;
                const uint4* pb = reinterpret_cast<const uint4*>(
                    W1s + ((size_t)nt * KT1 + kt) * 512 + lane * 16);
                bfr.u[0] = pb[0];
                bfr.u[1] = pb[1];
                acc[nti * 2 + 0] = __builtin_amdgcn_wmma_f32_16x16x32_bf16(
                    false, a0.v, false, bfr.v, (short)0, acc[nti * 2 + 0], false, false);
                acc[nti * 2 + 1] = __builtin_amdgcn_wmma_f32_16x16x32_bf16(
                    false, a1.v, false, bfr.v, (short)0, acc[nti * 2 + 1], false, false);
            }
        }
        #pragma unroll
        for (int nti = 0; nti < 4; ++nti) {
            const int nt = wave * 4 + nti;
            const float b1v = b1[nt * 16 + idx];
            #pragma unroll
            for (int i = 0; i < 8; ++i) {
                float h0 = acc[nti * 2 + 0][i] + b1v;
                float h1 = acc[nti * 2 + 1][i] + b1v;
                h0 = (h0 > 0.0f) ? h0 : 0.0f;
                h1 = (h1 > 0.0f) ? h1 : 0.0f;
                sH[(half * 8 + i) * DHID + nt * 16 + idx]      = (unsigned short)f2bf(h0);
                sH[(16 + half * 8 + i) * DHID + nt * 16 + idx] = (unsigned short)f2bf(h1);
            }
        }
    }
    __syncthreads();

    // ---- Stage 2: gate = sigmoid(h @ W2 + b2); gated = feats*gate in LDS ----
    // Two passes of 4 n-tiles to keep accumulators at 64 VGPRs.
    for (int pass = 0; pass < 2; ++pass) {
        v8f acc[8] = {};                                 // [nti*2 + mtile]
        for (int kt = 0; kt < KT2; ++kt) {
            Frag a0, a1;
            const uint4* pa0 =
                reinterpret_cast<const uint4*>(&sH[idx * DHID + kt * 32 + half * 8]);
            a0.u[0] = pa0[0];
            a0.u[1] = pa0[2];
            const uint4* pa1 = reinterpret_cast<const uint4*>(
                &sH[(16 + idx) * DHID + kt * 32 + half * 8]);
            a1.u[0] = pa1[0];
            a1.u[1] = pa1[2];
            #pragma unroll
            for (int nti = 0; nti < 4; ++nti) {
                const int nt = wave * 8 + pass * 4 + nti;   // 0..63
                Frag bfr;
                const uint4* pb = reinterpret_cast<const uint4*>(
                    W2s + ((size_t)nt * KT2 + kt) * 512 + lane * 16);
                bfr.u[0] = pb[0];
                bfr.u[1] = pb[1];
                acc[nti * 2 + 0] = __builtin_amdgcn_wmma_f32_16x16x32_bf16(
                    false, a0.v, false, bfr.v, (short)0, acc[nti * 2 + 0], false, false);
                acc[nti * 2 + 1] = __builtin_amdgcn_wmma_f32_16x16x32_bf16(
                    false, a1.v, false, bfr.v, (short)0, acc[nti * 2 + 1], false, false);
            }
        }
        #pragma unroll
        for (int nti = 0; nti < 4; ++nti) {
            const int nt = wave * 8 + pass * 4 + nti;
            const float b2v = b2[nt * 16 + idx];
            #pragma unroll
            for (int i = 0; i < 8; ++i) {
                const int n  = nt * 16 + idx;
                const int m0 = half * 8 + i;
                const int m1 = 16 + half * 8 + i;
                const float g0 = 1.0f / (1.0f + __expf(-(acc[nti * 2 + 0][i] + b2v)));
                const float g1 = 1.0f / (1.0f + __expf(-(acc[nti * 2 + 1][i] + b2v)));
                sGated[m0 * DIN + n] *= g0;    // in-place: f32 features -> gated
                sGated[m1 * DIN + n] *= g1;
            }
        }
    }
    __syncthreads();

    // ---- Stage 3: per-row top-256 of 1024 by |gated| (one wave per 4 rows) --
    for (int rr = 0; rr < 4; ++rr) {
        const int r = wave * 4 + rr;
        float v[32];
        #pragma unroll
        for (int j = 0; j < 32; ++j)
            v[j] = sGated[r * DIN + j * 32 + lane];   // lane holds elem 32j+lane

        // MSB-first binary search for the 256th-largest |x| bit pattern.
        unsigned int t = 0;
        for (int bit = 30; bit >= 0; --bit) {
            const unsigned int cand = t | (1u << bit);
            int c = 0;
            #pragma unroll
            for (int j = 0; j < 32; ++j)
                c += ((__float_as_uint(v[j]) & 0x7FFFFFFFu) >= cand) ? 1 : 0;
            for (int off = 16; off > 0; off >>= 1)
                c += __shfl_xor(c, off, 32);
            if (c >= KSEL) t = cand;
        }

        float* orow = out + (size_t)(rowBase + r) * KSEL;
        const unsigned int lmask = (1u << lane) - 1u;
        int base = 0;
        // Pass 1: strictly greater than threshold (count < 256 guaranteed).
        #pragma unroll
        for (int j = 0; j < 32; ++j) {
            const unsigned int key = __float_as_uint(v[j]) & 0x7FFFFFFFu;
            const bool p = key > t;
            const unsigned int m = (unsigned int)__ballot(p);
            const int pos = base + __popc(m & lmask);
            if (p) orow[pos] = v[j];
            base += __popc(m);
        }
        // Pass 2: ties at threshold fill the remaining slots.
        for (int j = 0; j < 32 && base < KSEL; ++j) {
            const unsigned int key = __float_as_uint(v[j]) & 0x7FFFFFFFu;
            const bool p = (key == t);
            const unsigned int m = (unsigned int)__ballot(p);
            const int pos = base + __popc(m & lmask);
            if (p && pos < KSEL) orow[pos] = v[j];
            base += __popc(m);
        }
    }
}

// ---------------------------------------------------------------------------
extern "C" void kernel_launch(void* const* d_in, const int* in_sizes, int n_in,
                              void* d_out, int out_size, void* d_ws, size_t ws_size,
                              hipStream_t stream) {
    const float* feats = (const float*)d_in[0];
    // d_in[1] = importance_weights: unused by the reference computation.
    const float* W1 = (const float*)d_in[2];
    const float* b1 = (const float*)d_in[3];
    const float* W2 = (const float*)d_in[4];
    const float* b2 = (const float*)d_in[5];
    float* out = (float*)d_out;

    const int N = in_sizes[0] / DIN;   // 131072

    unsigned short* W1s = (unsigned short*)d_ws;                 // 512K bf16 = 1 MB
    unsigned short* W2s = W1s + (size_t)DIN * DHID;              // 512K bf16 = 1 MB

    swizzle_weights_kernel<<<2048, 256, 0, stream>>>(W1, W2, W1s, W2s);
    thalamic_gate_topk_kernel<<<N / ROWS, 256, 0, stream>>>(feats, b1, b2,
                                                            W1s, W2s, out);
}
